// DecoderWithAttention_22101901705519
// MI455X (gfx1250) — matrix operand order repaired and checked by reading, not compiled, not run on hardware
//
#include <hip/hip_runtime.h>
#include <hip/hip_bf16.h>
#include <cstdint>
#include <cstddef>

// ---------------------------------------------------------------------------
// Problem constants (from the reference)
// ---------------------------------------------------------------------------
#define BB   32
#define NN   196
#define ENC  512
#define ATT  512
#define DEC  512
#define EMB  256
#define VV   10000
#define LL   80
#define TT   79          // L - 1
#define XDIM 768         // EMB + ENC
#define G3   1536        // 3 * DEC
#define MROWS (BB * NN)  // 6272 rows of att1

typedef __attribute__((ext_vector_type(16))) __bf16 v16bf;
typedef __attribute__((ext_vector_type(8)))  float  v8f;

// ---------------------------------------------------------------------------
// Scalar helpers
// ---------------------------------------------------------------------------
static __device__ __forceinline__ unsigned short f32_to_bf16_rne(float f) {
    union { float f; unsigned int u; } c; c.f = f;
    unsigned int u = c.u;
    unsigned int lsb = (u >> 16) & 1u;
    u += 0x7fffu + lsb;                 // round to nearest even
    return (unsigned short)(u >> 16);
}

static __device__ __forceinline__ __bf16 ushort_as_bf16(unsigned short u) {
    union { unsigned short u; __bf16 b; } c; c.u = u;
    return c.b;
}

static __device__ __forceinline__ float sigmoidf_(float x) {
    return 1.0f / (1.0f + expf(-x));
}

// ---------------------------------------------------------------------------
// WMMA row-of-tiles:  NT independent 16x16 f32 accumulators sharing one
// A fragment per k-block (A row-major, pre-offset to tile row; Bt
// column-major Bt[n*ldb + k], pre-offset to first of NT tile columns).
//
// Lane layouts per CDNA5 ISA 7.12.2:
//   A 16x32 bf16 : m = lane&15, kh = lane>>4, VGPR v holds K = 2v+(v>=4?8:0)+8kh, +1
//   B 32x16 bf16 : n = lane&15, kh = lane>>4, VGPR v holds K = 2v+(v>=4?8:0)+8kh, +1
//   C/D 16x16 f32: n = lane&15, VGPR v holds row M = v + 8*(lane>>4)
//
// NT independent accumulator chains -> NT-way ILP on the matrix pipe and
// NT-fold reuse of the A fragment (one A load feeds NT WMMAs).
// ---------------------------------------------------------------------------
template <int NT>
static __device__ __forceinline__ void
wmma_rowtile_bf16(const unsigned short* __restrict__ A, int lda,
                  const unsigned short* __restrict__ Bt, int ldb,
                  int kblocks, int lane, v8f acc[NT]) {
    const int mn = lane & 15;
    const int kh = lane >> 4;
    for (int kb = 0; kb < kblocks; ++kb) {
        v16bf a;
        const unsigned short* arow = A + mn * lda + kb * 32;
        #pragma unroll
        for (int v = 0; v < 8; ++v) {
            int k0 = 2 * v + ((v >= 4) ? 8 : 0) + 8 * kh;
            a[2 * v]     = ushort_as_bf16(arow[k0]);
            a[2 * v + 1] = ushort_as_bf16(arow[k0 + 1]);
        }
        #pragma unroll
        for (int j = 0; j < NT; ++j) {
            v16bf b;
            const unsigned short* bcol =
                Bt + (size_t)(j * 16 + mn) * ldb + kb * 32 + 8 * kh;
            #pragma unroll
            for (int v = 0; v < 16; ++v) b[v] = ushort_as_bf16(bcol[v + ((v >= 8) ? 8 : 0)]);
            acc[j] = __builtin_amdgcn_wmma_f32_16x16x32_bf16(
                         false, a, false, b, (short)0, acc[j], false, false);
        }
    }
}

// ---------------------------------------------------------------------------
// One-time kernels
// ---------------------------------------------------------------------------

// Stable descending argsort of caption_lengths (rank method) + emit caps /
// dec_lens / order output regions (as float, tuple concat layout).
__global__ void sort_kernel(const int* __restrict__ lens,
                            const int* __restrict__ caps,
                            int* __restrict__ order_ws,
                            int* __restrict__ declens_ws,
                            float* __restrict__ out_caps,
                            float* __restrict__ out_declens,
                            float* __restrict__ out_order) {
    __shared__ int ord_s[BB];
    int tid = threadIdx.x;
    if (tid < BB) {
        int li = lens[tid];
        int r = 0;
        for (int j = 0; j < BB; ++j) {
            int lj = lens[j];
            if (lj > li || (lj == li && j < tid)) r++;
        }
        ord_s[r] = tid;
    }
    __syncthreads();
    if (tid < BB) {
        int o = ord_s[tid];
        order_ws[tid]   = o;
        int dl          = lens[o] - 1;
        declens_ws[tid] = dl;
        out_declens[tid] = (float)dl;
        out_order[tid]   = (float)o;
    }
    __syncthreads();
    for (int i = tid; i < BB * LL; i += blockDim.x) {
        int b = i / LL, l = i % LL;
        out_caps[i] = (float)caps[ord_s[b] * LL + l];
    }
}

// Gather encoder_out by order and convert to bf16 (row-major BB*NN x ENC).
__global__ void pack_enc_kernel(const float* __restrict__ enc_all,
                                const int* __restrict__ order,
                                unsigned short* __restrict__ dst) {
    int i = blockIdx.x * blockDim.x + threadIdx.x;
    const int per_b = NN * ENC;
    if (i >= BB * per_b) return;
    int b = i / per_b;
    int rest = i - b * per_b;
    dst[i] = f32_to_bf16_rne(enc_all[(size_t)order[b] * per_b + rest]);
}

// f32 -> bf16, same layout.
__global__ void cvt_direct_kernel(const float* __restrict__ src,
                                  unsigned short* __restrict__ dst, int n) {
    int i = blockIdx.x * blockDim.x + threadIdx.x;
    if (i < n) dst[i] = f32_to_bf16_rne(src[i]);
}

// f32 (rows x cols, row-major) -> bf16 column-major dst[c*rows + r].
__global__ void cvt_transpose_kernel(const float* __restrict__ src,
                                     unsigned short* __restrict__ dst,
                                     int rows, int cols) {
    int i = blockIdx.x * blockDim.x + threadIdx.x;
    if (i >= rows * cols) return;
    int r = i / cols, c = i - r * cols;
    dst[(size_t)c * rows + r] = f32_to_bf16_rne(src[i]);
}

// h0 = mean_N(enc_sorted) @ W_init_h + b_init_h  (one block per batch elem)
__global__ void h0_kernel(const float* __restrict__ enc_all,
                          const int* __restrict__ order,
                          const float* __restrict__ Wi,
                          const float* __restrict__ bi,
                          float* __restrict__ h) {
    int b = blockIdx.x, tid = threadIdx.x;
    __shared__ float mean_s[ENC];
    const float* encb = enc_all + (size_t)order[b] * NN * ENC;
    for (int k = tid; k < ENC; k += blockDim.x) {
        float s = 0.f;
        for (int n = 0; n < NN; ++n) s += encb[(size_t)n * ENC + k];
        mean_s[k] = s * (1.0f / (float)NN);
    }
    __syncthreads();
    for (int j = tid; j < DEC; j += blockDim.x) {
        float acc = bi[j];
        for (int k = 0; k < ENC; ++k) acc += mean_s[k] * Wi[(size_t)k * DEC + j];
        h[(size_t)b * DEC + j] = acc;
    }
}

// att1 = enc_bf16 @ Wt_enc_att + b_enc_att   (M=6272, N=512, K=512)
// 8 waves/block; each wave: 1 M-tile x 4 N-tiles. 392*8 = 3136 waves exactly.
__global__ void att1_gemm_kernel(const unsigned short* __restrict__ encb,
                                 const unsigned short* __restrict__ wt,
                                 const float* __restrict__ bias,
                                 float* __restrict__ att1) {
    int wave = blockIdx.x * (blockDim.x >> 5) + (threadIdx.x >> 5);
    int mt  = wave >> 3;          // /8   (0..391)
    int grp = wave & 7;           // %8   -> 4 N-tiles each
    int nt0 = grp * 4;
    int lane = threadIdx.x & 31;
    v8f acc[4];
    #pragma unroll
    for (int j = 0; j < 4; ++j) acc[j] = (v8f){0.f,0.f,0.f,0.f,0.f,0.f,0.f,0.f};
    wmma_rowtile_bf16<4>(encb + (size_t)mt * 16 * ENC, ENC,
                         wt   + (size_t)nt0 * 16 * ENC, ENC,
                         ENC / 32, lane, acc);
    int n = lane & 15, mh = lane >> 4;
    #pragma unroll
    for (int j = 0; j < 4; ++j) {
        int col = (nt0 + j) * 16 + n;
        float bv = bias[col];
        #pragma unroll
        for (int v = 0; v < 8; ++v) {
            int row = mt * 16 + v + 8 * mh;
            att1[(size_t)row * ATT + col] = acc[j][v] + bv;
        }
    }
}

// ---------------------------------------------------------------------------
// Per-step kernels
// ---------------------------------------------------------------------------

// Attention + gate, block per batch element (256 threads).
__global__ void step_attn_kernel(const float* __restrict__ att1,
                                 const float* __restrict__ h,
                                 const float* __restrict__ Wd,
                                 const float* __restrict__ bd,
                                 const float* __restrict__ wfull,
                                 const float* __restrict__ bfull,
                                 const float* __restrict__ enc_all,
                                 const int* __restrict__ order,
                                 const float* __restrict__ emb,
                                 const int* __restrict__ caps,
                                 const float* __restrict__ Wfb,
                                 const float* __restrict__ bfb,
                                 const int* __restrict__ dec_lens,
                                 int t,
                                 float* __restrict__ alphas_out,
                                 unsigned short* __restrict__ x_bf,
                                 unsigned short* __restrict__ h_bf) {
    int b = blockIdx.x, tid = threadIdx.x;
    __shared__ float h_s[DEC];
    __shared__ float att2_s[ATT];
    __shared__ float alpha_s[NN];
    __shared__ float red[256];

    // load h row, also publish bf16 copy for the GRU hidden GEMM
    for (int k = tid; k < DEC; k += 256) {
        float hv = h[(size_t)b * DEC + k];
        h_s[k] = hv;
        h_bf[(size_t)b * DEC + k] = f32_to_bf16_rne(hv);
    }
    __syncthreads();

    // att2 = h @ W_dec_att + b_dec_att  (coalesced over j)
    for (int j = tid; j < ATT; j += 256) {
        float acc = bd[j];
        for (int k = 0; k < DEC; ++k) acc += h_s[k] * Wd[(size_t)k * ATT + j];
        att2_s[j] = acc;
    }
    __syncthreads();

    // e[n] = relu(att1[b,n,:] + att2) . w_full + b_full
    float e = -3.402823466e38f;
    if (tid < NN) {
        const float* arow = att1 + ((size_t)b * NN + tid) * ATT;
        float acc = bfull[0];
        for (int j = 0; j < ATT; ++j) {
            float r = arow[j] + att2_s[j];
            r = r > 0.f ? r : 0.f;
            acc += r * wfull[j];
        }
        e = acc;
    }
    // softmax over N=196
    red[tid] = e;
    __syncthreads();
    for (int s = 128; s > 0; s >>= 1) {
        if (tid < s) red[tid] = fmaxf(red[tid], red[tid + s]);
        __syncthreads();
    }
    float maxv = red[0];
    __syncthreads();
    float p = (tid < NN) ? expf(e - maxv) : 0.f;
    red[tid] = p;
    __syncthreads();
    for (int s = 128; s > 0; s >>= 1) {
        if (tid < s) red[tid] += red[tid + s];
        __syncthreads();
    }
    float inv_sum = 1.0f / red[0];
    bool active = (t < dec_lens[b]);
    if (tid < NN) {
        float a = p * inv_sum;
        alpha_s[tid] = a;
        alphas_out[((size_t)b * TT + t) * NN + tid] = active ? a : 0.f;
    }
    __syncthreads();

    // awe (context), gate, and pack x = [emb_t, gate*awe] as bf16
    const float* encb = enc_all + (size_t)order[b] * NN * ENC;
    int cap = caps[order[b] * LL + t];
    const float* erow = emb + (size_t)cap * EMB;
    for (int d = tid; d < ENC; d += 256) {
        float aw = 0.f;
        for (int n = 0; n < NN; ++n) aw += alpha_s[n] * encb[(size_t)n * ENC + d];
        float g = bfb[d];
        for (int k = 0; k < DEC; ++k) g += h_s[k] * Wfb[(size_t)k * ENC + d];
        g = sigmoidf_(g);
        x_bf[(size_t)b * XDIM + EMB + d] = f32_to_bf16_rne(g * aw);
    }
    for (int d = tid; d < EMB; d += 256) {
        x_bf[(size_t)b * XDIM + d] = f32_to_bf16_rne(erow[d]);
    }
}

// gi = x @ W_ih^T + b_ih  and  gh = h @ W_hh^T + b_hh  (WMMA)
// One wave per block (max WGP spread). 96 waves: [0,48)=gi, [48,96)=gh,
// each wave: 1 M-tile x 4 N-tiles (96 N-tiles / 4 = 24 groups x 2 M-tiles).
__global__ void gru_gemm_kernel(const unsigned short* __restrict__ x_bf,
                                const unsigned short* __restrict__ h_bf,
                                const unsigned short* __restrict__ wih,
                                const unsigned short* __restrict__ whh,
                                const float* __restrict__ b_ih,
                                const float* __restrict__ b_hh,
                                float* __restrict__ gi,
                                float* __restrict__ gh) {
    int wave = blockIdx.x;
    int which = (wave >= 48) ? 1 : 0;
    int w2 = wave - which * 48;
    int mt  = w2 / 24;
    int grp = w2 % 24;
    int nt0 = grp * 4;
    int lane = threadIdx.x & 31;
    v8f acc[4];
    #pragma unroll
    for (int j = 0; j < 4; ++j) acc[j] = (v8f){0.f,0.f,0.f,0.f,0.f,0.f,0.f,0.f};
    const float* bias;
    float* out;
    if (which == 0) {
        wmma_rowtile_bf16<4>(x_bf + (size_t)mt * 16 * XDIM, XDIM,
                             wih  + (size_t)nt0 * 16 * XDIM, XDIM,
                             XDIM / 32, lane, acc);
        bias = b_ih; out = gi;
    } else {
        wmma_rowtile_bf16<4>(h_bf + (size_t)mt * 16 * DEC, DEC,
                             whh  + (size_t)nt0 * 16 * DEC, DEC,
                             DEC / 32, lane, acc);
        bias = b_hh; out = gh;
    }
    int n = lane & 15, mh = lane >> 4;
    #pragma unroll
    for (int j = 0; j < 4; ++j) {
        int col = (nt0 + j) * 16 + n;
        float bv = bias[col];
        #pragma unroll
        for (int v = 0; v < 8; ++v) {
            int row = mt * 16 + v + 8 * mh;
            out[(size_t)row * G3 + col] = acc[j][v] + bv;
        }
    }
}

// GRU pointwise update; writes carried h (masked) and h_new (bf16) for FC.
__global__ void gru_update_kernel(const float* __restrict__ gi,
                                  const float* __restrict__ gh,
                                  const int* __restrict__ dec_lens,
                                  int t,
                                  float* __restrict__ h,
                                  unsigned short* __restrict__ hn_bf) {
    int idx = blockIdx.x * blockDim.x + threadIdx.x;
    if (idx >= BB * DEC) return;
    int b = idx / DEC, j = idx - b * DEC;
    const float* gib = gi + (size_t)b * G3;
    const float* ghb = gh + (size_t)b * G3;
    float r  = sigmoidf_(gib[j] + ghb[j]);
    float z  = sigmoidf_(gib[DEC + j] + ghb[DEC + j]);
    float nn = tanhf(gib[2 * DEC + j] + r * ghb[2 * DEC + j]);
    float hv = h[idx];
    float hn = (1.0f - z) * nn + z * hv;
    bool active = (t < dec_lens[b]);
    h[idx] = active ? hn : hv;            // carried state
    hn_bf[idx] = f32_to_bf16_rne(hn);     // preds always use h_new
}

// preds = h_new @ W_fc + b_fc, masked store into output (WMMA)
// One wave per block; each wave: 1 M-tile x 5 N-tiles (625 = 125 * 5).
// 2 * 125 = 250 waves exactly; 5 independent accumulator chains.
__global__ void fc_gemm_kernel(const unsigned short* __restrict__ hn_bf,
                               const unsigned short* __restrict__ wfct,
                               const float* __restrict__ b_fc,
                               const int* __restrict__ dec_lens,
                               int t,
                               float* __restrict__ preds) {
    int wave = blockIdx.x;
    int mt  = wave / 125;
    int grp = wave % 125;
    int nt0 = grp * 5;
    int lane = threadIdx.x & 31;
    v8f acc[5];
    #pragma unroll
    for (int j = 0; j < 5; ++j) acc[j] = (v8f){0.f,0.f,0.f,0.f,0.f,0.f,0.f,0.f};
    wmma_rowtile_bf16<5>(hn_bf + (size_t)mt * 16 * DEC, DEC,
                         wfct  + (size_t)nt0 * 16 * DEC, DEC,
                         DEC / 32, lane, acc);
    int n = lane & 15, mh = lane >> 4;
    #pragma unroll
    for (int j = 0; j < 5; ++j) {
        int col = (nt0 + j) * 16 + n;
        float bv = b_fc[col];
        #pragma unroll
        for (int v = 0; v < 8; ++v) {
            int row = mt * 16 + v + 8 * mh;              // batch index 0..31
            bool active = (t < dec_lens[row]);
            preds[((size_t)row * TT + t) * VV + col] = active ? (acc[j][v] + bv) : 0.f;
        }
    }
}

// ---------------------------------------------------------------------------
// Host launch
// ---------------------------------------------------------------------------
extern "C" void kernel_launch(void* const* d_in, const int* in_sizes, int n_in,
                              void* d_out, int out_size, void* d_ws, size_t ws_size,
                              hipStream_t stream) {
    (void)in_sizes; (void)n_in; (void)out_size; (void)ws_size;

    const float* encoder_out = (const float*)d_in[0];
    const int*   enc_caps    = (const int*)d_in[1];
    const int*   cap_lens    = (const int*)d_in[2];
    const float* W_enc_att   = (const float*)d_in[3];
    const float* b_enc_att   = (const float*)d_in[4];
    const float* W_dec_att   = (const float*)d_in[5];
    const float* b_dec_att   = (const float*)d_in[6];
    const float* w_full      = (const float*)d_in[7];
    const float* b_full      = (const float*)d_in[8];
    const float* emb         = (const float*)d_in[9];
    const float* W_ih        = (const float*)d_in[10];
    const float* W_hh        = (const float*)d_in[11];
    const float* b_ih        = (const float*)d_in[12];
    const float* b_hh        = (const float*)d_in[13];
    const float* W_init_h    = (const float*)d_in[14];
    const float* b_init_h    = (const float*)d_in[15];
    const float* W_fbeta     = (const float*)d_in[16];
    const float* b_fbeta     = (const float*)d_in[17];
    const float* W_fc        = (const float*)d_in[18];
    const float* b_fc        = (const float*)d_in[19];

    // ---- workspace carve (aligned 256B), ~35 MB total -------------------
    char* w = (char*)d_ws;
    size_t off = 0;
    auto carve = [&](size_t bytes) -> void* {
        void* p = w + off;
        off = (off + bytes + 255) & ~(size_t)255;
        return p;
    };
    int*            order_ws   = (int*)carve(BB * 4);
    int*            declens_ws = (int*)carve(BB * 4);
    float*          h_ws       = (float*)carve((size_t)BB * DEC * 4);
    unsigned short* h_bf       = (unsigned short*)carve((size_t)BB * DEC * 2);
    unsigned short* hn_bf      = (unsigned short*)carve((size_t)BB * DEC * 2);
    unsigned short* x_bf       = (unsigned short*)carve((size_t)BB * XDIM * 2);
    float*          gi_ws      = (float*)carve((size_t)BB * G3 * 4);
    float*          gh_ws      = (float*)carve((size_t)BB * G3 * 4);
    unsigned short* enc_bf     = (unsigned short*)carve((size_t)MROWS * ENC * 2);
    float*          att1_ws    = (float*)carve((size_t)MROWS * ATT * 4);
    unsigned short* wt_enc     = (unsigned short*)carve((size_t)ENC * ATT * 2);
    unsigned short* wih_bf     = (unsigned short*)carve((size_t)G3 * XDIM * 2);
    unsigned short* whh_bf     = (unsigned short*)carve((size_t)G3 * DEC * 2);
    unsigned short* wfc_t      = (unsigned short*)carve((size_t)VV * DEC * 2);

    // ---- output layout (tuple concat, float) ----------------------------
    float* out       = (float*)d_out;
    float* out_pred  = out;                                        // 32*79*10000
    float* out_caps  = out + (size_t)BB * TT * VV;                 // 32*80
    float* out_dl    = out_caps + (size_t)BB * LL;                 // 32
    float* out_alpha = out_dl + BB;                                // 32*79*196
    float* out_order = out_alpha + (size_t)BB * TT * NN;           // 32

    // ---- one-time setup -------------------------------------------------
    sort_kernel<<<1, 256, 0, stream>>>(cap_lens, enc_caps, order_ws, declens_ws,
                                       out_caps, out_dl, out_order);

    {
        int n = BB * NN * ENC;
        pack_enc_kernel<<<(n + 255) / 256, 256, 0, stream>>>(encoder_out, order_ws, enc_bf);
    }
    cvt_transpose_kernel<<<(ENC * ATT + 255) / 256, 256, 0, stream>>>(W_enc_att, wt_enc, ENC, ATT);
    cvt_direct_kernel<<<(G3 * XDIM + 255) / 256, 256, 0, stream>>>(W_ih, wih_bf, G3 * XDIM);
    cvt_direct_kernel<<<(G3 * DEC + 255) / 256, 256, 0, stream>>>(W_hh, whh_bf, G3 * DEC);
    cvt_transpose_kernel<<<(DEC * VV + 255) / 256, 256, 0, stream>>>(W_fc, wfc_t, DEC, VV);

    h0_kernel<<<BB, 256, 0, stream>>>(encoder_out, order_ws, W_init_h, b_init_h, h_ws);

    {
        // 392 M-tiles x 8 groups(4 N-tiles) = 3136 waves, 8 waves/block
        att1_gemm_kernel<<<392, 256, 0, stream>>>(enc_bf, wt_enc, b_enc_att, att1_ws);
    }

    // ---- sequential decode ---------------------------------------------
    for (int t = 0; t < TT; ++t) {
        step_attn_kernel<<<BB, 256, 0, stream>>>(
            att1_ws, h_ws, W_dec_att, b_dec_att, w_full, b_full,
            encoder_out, order_ws, emb, enc_caps, W_fbeta, b_fbeta,
            declens_ws, t, out_alpha, x_bf, h_bf);

        // 96 one-wave blocks spread across WGPs
        gru_gemm_kernel<<<96, 32, 0, stream>>>(
            x_bf, h_bf, wih_bf, whh_bf, b_ih, b_hh, gi_ws, gh_ws);

        gru_update_kernel<<<(BB * DEC + 255) / 256, 256, 0, stream>>>(
            gi_ws, gh_ws, declens_ws, t, h_ws, hn_bf);

        // 250 one-wave blocks, 5 accumulator chains each
        fc_gemm_kernel<<<250, 32, 0, stream>>>(
            hn_bf, wfc_t, b_fc, declens_ws, t, out_pred);
    }
}
